// MultiHeadAttention_40604620816912
// MI455X (gfx1250) — matrix operand order repaired and checked by reading, not compiled, not run on hardware
//
#include <hip/hip_runtime.h>
#include <hip/hip_bf16.h>
#include <cstdint>

#define BB   4
#define SS   2048
#define DD   1024
#define HH   16
#define DHH  64
// BS = BB*SS = 8192

typedef __attribute__((ext_vector_type(16))) __bf16       v16bf;
typedef __attribute__((ext_vector_type(8)))  float        v8f;
typedef __attribute__((ext_vector_type(4)))  unsigned int u32x4;
typedef __attribute__((ext_vector_type(8)))  int          i32x8;
typedef __attribute__((ext_vector_type(4)))  int          i32x4;

union OpBf {
  v16bf v;
  u32x4 q[2];
  unsigned short h[16];
};

__device__ __forceinline__ unsigned short f32_bf16(float f) {
  unsigned int x = __float_as_uint(f);
  x += 0x7FFFu + ((x >> 16) & 1u);          // round-to-nearest-even
  return (unsigned short)(x >> 16);
}

__device__ __forceinline__ v8f wmma_bf16(const OpBf& a, const OpBf& b, v8f c) {
  return __builtin_amdgcn_wmma_f32_16x16x32_bf16(
      /*neg_a=*/false, a.v, /*neg_b=*/false, b.v,
      /*c_mod=*/(short)0, c, /*reuse_a=*/false, /*reuse_b=*/false);
}

#if defined(__has_builtin)
#if __has_builtin(__builtin_amdgcn_tensor_load_to_lds) && \
    __has_builtin(__builtin_amdgcn_s_wait_tensorcnt)
#define HAVE_TDM 1
#endif
#endif

#ifdef HAVE_TDM
// ---------------------------------------------------------------------------
// TDM: async 2D tile (rows x cols, bf16) from global -> LDS via D# descriptor.
// Bit packing per CDNA5 ISA ch.8 (group0 128b, group1 256b); groups 2/3 NULL
// (2D tensor). Tracked on TENSORcnt.
// ---------------------------------------------------------------------------
__device__ __forceinline__ void tdm_load_2d(
    unsigned lds_addr, const void* gaddr,
    unsigned tile_x, unsigned tile_y,            // elements / rows
    unsigned tensor_x, unsigned tensor_y,        // tensor extents (elements)
    unsigned stride_x) {                         // row stride (elements)
  unsigned long long ga = (unsigned long long)(uintptr_t)gaddr;
  u32x4 g0;
  g0[0] = 1u;                                   // count=1, user descriptor
  g0[1] = lds_addr;                             // LDS byte address
  g0[2] = (unsigned)ga;                         // global_addr[31:0]
  g0[3] = (unsigned)(ga >> 32) | (2u << 30);    // global_addr[56:32] | type=2
  i32x8 g1;
  g1[0] = (int)(1u << 16);                      // workgroup_mask=0, data_size=1 (2B)
  g1[1] = (int)((tensor_x & 0xFFFFu) << 16);    // tensor_dim0[15:0]
  g1[2] = (int)(((tensor_x >> 16) & 0xFFFFu) | ((tensor_y & 0xFFFFu) << 16));
  g1[3] = (int)(((tensor_y >> 16) & 0xFFFFu) | ((tile_x & 0xFFFFu) << 16));
  g1[4] = (int)(tile_y & 0xFFFFu);              // tile_dim1; tile_dim2=0
  g1[5] = (int)stride_x;                        // tensor_dim0_stride[31:0]
  g1[6] = 0;                                    // stride0 hi, stride1 lo
  g1[7] = 0;                                    // stride1 hi (unused, 2D)
  i32x4 z4 = {0, 0, 0, 0};
#if __clang_major__ >= 23
  i32x8 z8 = {0, 0, 0, 0, 0, 0, 0, 0};
  __builtin_amdgcn_tensor_load_to_lds(g0, g1, z4, z4, z8, 0);
#else
  __builtin_amdgcn_tensor_load_to_lds(g0, g1, z4, z4, 0);
#endif
}

__device__ __forceinline__ unsigned lds_byte_addr(const void* p) {
  return (unsigned)(uintptr_t)p;                // generic->LDS: addr[31:0]
}
#endif  // HAVE_TDM

// ---------------------------------------------------------------------------
// fp32 -> bf16 bulk convert (8 elements / thread)
// ---------------------------------------------------------------------------
__global__ __launch_bounds__(256) void cvt_f32_bf16(
    const float* __restrict__ in, unsigned short* __restrict__ out, int n8) {
  int i = blockIdx.x * blockDim.x + threadIdx.x;
  if (i >= n8) return;
  size_t base = (size_t)i * 8;
  float4 a = *(const float4*)(in + base);
  float4 b = *(const float4*)(in + base + 4);
  u32x4 o;
  o[0] = (unsigned)f32_bf16(a.x) | ((unsigned)f32_bf16(a.y) << 16);
  o[1] = (unsigned)f32_bf16(a.z) | ((unsigned)f32_bf16(a.w) << 16);
  o[2] = (unsigned)f32_bf16(b.x) | ((unsigned)f32_bf16(b.y) << 16);
  o[3] = (unsigned)f32_bf16(b.z) | ((unsigned)f32_bf16(b.w) << 16);
  *(u32x4*)(out + base) = o;
}

// ---------------------------------------------------------------------------
// fp32 weight [k][n] -> bf16 transposed Wt[n][k]
// ---------------------------------------------------------------------------
__global__ __launch_bounds__(256) void cvt_transpose_w(
    const float* __restrict__ W, unsigned short* __restrict__ Wt) {
  int i = blockIdx.x * blockDim.x + threadIdx.x;       // [0, D*D/8)
  int n  = i / (DD / 8);
  int k8 = (i % (DD / 8)) * 8;
  u32x4 o;
#pragma unroll
  for (int j = 0; j < 4; ++j) {
    unsigned short lo = f32_bf16(W[(size_t)(k8 + 2 * j)     * DD + n]);
    unsigned short hi = f32_bf16(W[(size_t)(k8 + 2 * j + 1) * DD + n]);
    o[j] = (unsigned)lo | ((unsigned)hi << 16);
  }
  *(u32x4*)(Wt + (size_t)n * DD + k8) = o;
}

// ---------------------------------------------------------------------------
// GEMM with register ping-pong double buffering so loads for chunk k+32
// overlap the WMMA chain on chunk k (avoids s_wait_loadcnt 0 serialization).
// mode 0: bf16 out [B,H,S,DH]; mode 1: bf16 out [B,H,DH,S]; mode 2: f32 [BS,D]
// ---------------------------------------------------------------------------
__device__ __forceinline__ void gemm_load_chunk(
    const unsigned short* arow, const unsigned short* const* brow,
    int k0, OpBf& a, OpBf* b) {
  a.q[0] = *(const u32x4*)(arow + k0);
  a.q[1] = *(const u32x4*)(arow + k0 + 16);
#pragma unroll
  for (int t = 0; t < 4; ++t) {
    b[t].q[0] = *(const u32x4*)(brow[t] + k0);
    b[t].q[1] = *(const u32x4*)(brow[t] + k0 + 16);
  }
}

__global__ __launch_bounds__(128) void gemm_bf16(
    const unsigned short* __restrict__ X,   // [8192, 1024] bf16
    const unsigned short* __restrict__ Wt,  // [1024(n), 1024(k)] bf16
    const float* __restrict__ bias,         // [1024] f32
    void* __restrict__ out, int mode) {
  const int lane = threadIdx.x & 31;
  const int wave = threadIdx.x >> 5;
  const int col  = lane & 15;
  const int hi   = lane >> 4;
  const int m0 = (blockIdx.x * 4 + wave) * 16;
  const int n0 = blockIdx.y * 64;

  v8f acc[4] = {};

  const unsigned short* arow = X + (size_t)(m0 + col) * DD + (hi ? 8 : 0);
  const unsigned short* brow[4];
#pragma unroll
  for (int t = 0; t < 4; ++t)
    brow[t] = Wt + (size_t)(n0 + t * 16 + col) * DD + (hi ? 8 : 0);

  OpBf aA, bA[4], aB, bB[4];
  gemm_load_chunk(arow, brow, 0, aA, bA);
  for (int k0 = 0; k0 < DD; k0 += 64) {
    gemm_load_chunk(arow, brow, k0 + 32, aB, bB);
#pragma unroll
    for (int t = 0; t < 4; ++t) acc[t] = wmma_bf16(aA, bA[t], acc[t]);
    if (k0 + 64 < DD) gemm_load_chunk(arow, brow, k0 + 64, aA, bA);
#pragma unroll
    for (int t = 0; t < 4; ++t) acc[t] = wmma_bf16(aB, bB[t], acc[t]);
  }

#pragma unroll
  for (int t = 0; t < 4; ++t) {
    int n = n0 + t * 16 + col;
    float bv = bias[n];
#pragma unroll
    for (int r = 0; r < 8; ++r) {
      int m = m0 + hi * 8 + r;
      float val = acc[t][r] + bv;
      if (mode == 2) {
        ((float*)out)[(size_t)m * DD + n] = val;
      } else {
        int b = m >> 11, s = m & (SS - 1);
        int h = n >> 6,  d = n & (DHH - 1);
        size_t idx = (mode == 0)
            ? ((size_t)(b * HH + h) * SS + s) * DHH + d
            : ((size_t)(b * HH + h) * DHH + d) * SS + s;
        ((unsigned short*)out)[idx] = f32_bf16(val);
      }
    }
  }
}

// ---------------------------------------------------------------------------
// Flash attention: one wave per (b, h, 16-row q-tile). K/V tiles streamed
// into double-buffered LDS by the Tensor Data Mover (TENSORcnt sync); WMMA
// operands read from LDS via b128 ds_loads. Online softmax in registers;
// P tile transposed C-layout -> A-layout through LDS.
// ---------------------------------------------------------------------------
__global__ __launch_bounds__(32) void flash_attn(
    const unsigned short* __restrict__ Q,    // [B,H,S,DH]  bf16
    const unsigned short* __restrict__ K,    // [B,H,S,DH]  bf16
    const unsigned short* __restrict__ Vt,   // [B,H,DH,S]  bf16
    const int* __restrict__ mask,            // [B,S]
    unsigned short* __restrict__ out) {      // [B,S,D]     bf16
  __shared__ unsigned short Kbuf[2][32 * DHH];   // 2 x 4 KB  [key][dim]
  __shared__ unsigned short Vbuf[2][DHH * 32];   // 2 x 4 KB  [dim][key]
  __shared__ unsigned short P[16 * 32];          // 1 KB probability tile

  const int lane = threadIdx.x & 31;
  const int col  = lane & 15;
  const int hi   = lane >> 4;
  const int qt = blockIdx.x, h = blockIdx.y, b = blockIdx.z;

  const size_t bh = (size_t)(b * HH + h);
  const unsigned short* Qp = Q  + (bh * SS + qt * 16) * DHH;
  const unsigned short* Kp = K  + bh * SS * DHH;
  const unsigned short* Vp = Vt + bh * DHH * SS;
  const int* mp = mask + (size_t)b * SS;

  // Q A-operands for the two k=32 dim-chunks (held for the whole kernel)
  OpBf qa[2];
  {
    const unsigned short* qr = Qp + col * DHH + (hi ? 8 : 0);
    qa[0].q[0] = *(const u32x4*)(qr);
    qa[0].q[1] = *(const u32x4*)(qr + 16);
    qa[1].q[0] = *(const u32x4*)(qr + 32);
    qa[1].q[1] = *(const u32x4*)(qr + 48);
  }

  v8f o[4] = {};
  float mrun[8], lrun[8];
#pragma unroll
  for (int r = 0; r < 8; ++r) { mrun[r] = -3.0e38f; lrun[r] = 0.f; }

#ifdef HAVE_TDM
  // Prime double buffer: K tile (32x64) and Vt tile (64x32) for block j=0.
  tdm_load_2d(lds_byte_addr(&Kbuf[0][0]), Kp, DHH, 32, DHH, SS, DHH);
  tdm_load_2d(lds_byte_addr(&Vbuf[0][0]), Vp, 32, DHH, SS, DHH, SS);
#endif
  int cur = 0;

  for (int j = 0; j < SS; j += 32) {
#ifdef HAVE_TDM
    if (j + 32 < SS) {
      tdm_load_2d(lds_byte_addr(&Kbuf[cur ^ 1][0]),
                  Kp + (size_t)(j + 32) * DHH, DHH, 32, DHH, SS, DHH);
      tdm_load_2d(lds_byte_addr(&Vbuf[cur ^ 1][0]),
                  Vp + (size_t)(j + 32), 32, DHH, SS, DHH, SS);
      __builtin_amdgcn_s_wait_tensorcnt(2);   // current buffer complete
    } else {
      __builtin_amdgcn_s_wait_tensorcnt(0);   // drain
    }
#else
    // Fallback: cooperative per-lane staging of this block's K/V tiles.
    {
      const unsigned short* kg = Kp + (size_t)j * DHH;   // 32 rows x 64
#pragma unroll
      for (int c = 0; c < 8; ++c) {               // 256 16B-chunks / 32 lanes
        int idx = c * 32 + lane;
        int r = idx >> 3, ofs = (idx & 7) * 8;
        *(u32x4*)(&Kbuf[0][r * DHH + ofs]) = *(const u32x4*)(kg + (size_t)r * DHH + ofs);
      }
      const unsigned short* vg = Vp + j;                 // 64 rows x 32
#pragma unroll
      for (int c = 0; c < 8; ++c) {
        int idx = c * 32 + lane;
        int r = idx >> 2, ofs = (idx & 3) * 8;
        *(u32x4*)(&Vbuf[0][r * 32 + ofs]) = *(const u32x4*)(vg + (size_t)r * SS + ofs);
      }
    }
    asm volatile("s_wait_dscnt 0" ::: "memory");
#endif

    // ---- scores: two 16-key tiles, accumulated over two 32-dim chunks
    v8f s0 = {}, s1 = {};
#pragma unroll
    for (int c = 0; c < 2; ++c) {
      const unsigned short* k0 = &Kbuf[cur][(col)      * DHH + c * 32 + (hi ? 8 : 0)];
      const unsigned short* k1 = &Kbuf[cur][(col + 16) * DHH + c * 32 + (hi ? 8 : 0)];
      OpBf b0, b1;
      b0.q[0] = *(const u32x4*)(k0);
      b0.q[1] = *(const u32x4*)(k0 + 16);
      b1.q[0] = *(const u32x4*)(k1);
      b1.q[1] = *(const u32x4*)(k1 + 16);
      s0 = wmma_bf16(qa[c], b0, s0);
      s1 = wmma_bf16(qa[c], b1, s1);
    }

    // ---- mask + scale + online softmax (row stats per 16-lane half)
    float pen0 = -1.0e9f * (float)mp[j + col];
    float pen1 = -1.0e9f * (float)mp[j + 16 + col];
#pragma unroll
    for (int r = 0; r < 8; ++r) {
      float a0 = s0[r] * 0.125f + pen0;          // 1/sqrt(64)
      float a1 = s1[r] * 0.125f + pen1;
      float mx = fmaxf(a0, a1);
      mx = fmaxf(mx, __shfl_xor(mx, 1));
      mx = fmaxf(mx, __shfl_xor(mx, 2));
      mx = fmaxf(mx, __shfl_xor(mx, 4));
      mx = fmaxf(mx, __shfl_xor(mx, 8));         // stays within 16-lane half
      float mnew  = fmaxf(mrun[r], mx);
      float alpha = __expf(mrun[r] - mnew);
      mrun[r] = mnew;
      float p0 = __expf(a0 - mnew);
      float p1 = __expf(a1 - mnew);
      float rs = p0 + p1;
      rs += __shfl_xor(rs, 1);
      rs += __shfl_xor(rs, 2);
      rs += __shfl_xor(rs, 4);
      rs += __shfl_xor(rs, 8);
      lrun[r] = lrun[r] * alpha + rs;
      o[0][r] *= alpha; o[1][r] *= alpha; o[2][r] *= alpha; o[3][r] *= alpha;
      s0[r] = p0; s1[r] = p1;
    }

    // ---- transpose P (C-layout, col-per-lane) -> A-layout via LDS
#pragma unroll
    for (int r = 0; r < 8; ++r) {
      P[(r + hi * 8) * 32 + col]      = f32_bf16(s0[r]);
      P[(r + hi * 8) * 32 + 16 + col] = f32_bf16(s1[r]);
    }
    asm volatile("s_wait_dscnt 0" ::: "memory");   // single wave, explicit anyway
    OpBf pa;
    const unsigned short* pr = &P[col * 32 + (hi ? 8 : 0)];
    pa.q[0] = *(const u32x4*)(pr);
    pa.q[1] = *(const u32x4*)(pr + 16);

    // ---- O += P @ V  from LDS Vbuf [dim][key]
#pragma unroll
    for (int nt = 0; nt < 4; ++nt) {
      const unsigned short* vr = &Vbuf[cur][(nt * 16 + col) * 32 + (hi ? 8 : 0)];
      OpBf vb;
      vb.q[0] = *(const u32x4*)(vr);
      vb.q[1] = *(const u32x4*)(vr + 16);
      o[nt] = wmma_bf16(pa, vb, o[nt]);
    }
#ifdef HAVE_TDM
    cur ^= 1;
#endif
  }

  // ---- normalize + store bf16 to [B,S,D]
  unsigned short* op = out + ((size_t)b * SS + (size_t)qt * 16) * DD + h * DHH;
#pragma unroll
  for (int r = 0; r < 8; ++r) {
    float inv = 1.0f / lrun[r];
    int row = r + hi * 8;
#pragma unroll
    for (int nt = 0; nt < 4; ++nt)
      op[(size_t)row * DD + nt * 16 + col] = f32_bf16(o[nt][r] * inv);
  }
}

// ---------------------------------------------------------------------------
// Host-side orchestration
// ---------------------------------------------------------------------------
extern "C" void kernel_launch(void* const* d_in, const int* in_sizes, int n_in,
                              void* d_out, int out_size, void* d_ws, size_t ws_size,
                              hipStream_t stream) {
  const float* queries = (const float*)d_in[0];
  const float* keys    = (const float*)d_in[1];
  const float* values  = (const float*)d_in[2];
  const int*   mask    = (const int*)  d_in[3];
  const float* wq = (const float*)d_in[4];
  const float* bq = (const float*)d_in[5];
  const float* wk = (const float*)d_in[6];
  const float* bk = (const float*)d_in[7];
  const float* wv = (const float*)d_in[8];
  const float* bv = (const float*)d_in[9];
  const float* wo = (const float*)d_in[10];
  const float* bo = (const float*)d_in[11];

  const size_t ACT = (size_t)BB * SS * DD;   // 8,388,608 elements
  const size_t WD  = (size_t)DD * DD;        // 1,048,576 elements

  unsigned short* ws  = (unsigned short*)d_ws;
  unsigned short* Xq  = ws;
  unsigned short* Xk  = Xq + ACT;
  unsigned short* Xv  = Xk + ACT;
  unsigned short* Wqt = Xv + ACT;
  unsigned short* Wkt = Wqt + WD;
  unsigned short* Wvt = Wkt + WD;
  unsigned short* Wot = Wvt + WD;
  unsigned short* Qp  = Wot + WD;            // [B,H,S,DH]
  unsigned short* Kp  = Qp + ACT;            // [B,H,S,DH]
  unsigned short* Vt  = Kp + ACT;            // [B,H,DH,S]
  unsigned short* AO  = Vt + ACT;            // [B,S,D]  attention output

  // 1) convert activations to bf16
  int n8 = (int)(ACT / 8);
  cvt_f32_bf16<<<n8 / 256, 256, 0, stream>>>(queries, Xq, n8);
  cvt_f32_bf16<<<n8 / 256, 256, 0, stream>>>(keys,    Xk, n8);
  cvt_f32_bf16<<<n8 / 256, 256, 0, stream>>>(values,  Xv, n8);

  // 2) convert + transpose weights to bf16
  int nw = (int)(WD / 8);
  cvt_transpose_w<<<nw / 256, 256, 0, stream>>>(wq, Wqt);
  cvt_transpose_w<<<nw / 256, 256, 0, stream>>>(wk, Wkt);
  cvt_transpose_w<<<nw / 256, 256, 0, stream>>>(wv, Wvt);
  cvt_transpose_w<<<nw / 256, 256, 0, stream>>>(wo, Wot);

  // 3) Q/K/V projections (V stored head-transposed)
  dim3 gg(BB * SS / 64, DD / 64);            // (128, 16), 128 threads/block
  gemm_bf16<<<gg, 128, 0, stream>>>(Xq, Wqt, bq, Qp, 0);
  gemm_bf16<<<gg, 128, 0, stream>>>(Xk, Wkt, bk, Kp, 0);
  gemm_bf16<<<gg, 128, 0, stream>>>(Xv, Wvt, bv, Vt, 1);

  // 4) flash attention: one wave per (b, h, 16-row q-tile)
  dim3 fg(SS / 16, HH, BB);                  // (128, 16, 4)
  flash_attn<<<fg, 32, 0, stream>>>(Qp, Kp, Vt, mask, AO);

  // 5) output projection -> f32 d_out
  gemm_bf16<<<gg, 128, 0, stream>>>(AO, Wot, bo, d_out, 2);
}
